// Attention_33311766348358
// MI455X (gfx1250) — matrix operand order repaired and checked by reading, not compiled, not run on hardware
//
#include <hip/hip_runtime.h>
#include <hip/hip_bf16.h>

// ---------------------------------------------------------------------------
// Bahdanau attention, MI455X (gfx1250, wave32, WMMA bf16).
//   context = (softmax(tanh(feat@W1 + hid@W2 + b) @ Wv)ᵀ · feat) @ W1 + b1
// Main GEMM (131072x1024 @ 1024x1024) on v_wmma_f32_16x16x32_bf16.
// W1 is pre-packed into per-lane WMMA fragment layout so B fragments are
// direct L2-resident global_load_b128s; only A (streamed once) uses LDS.
// ---------------------------------------------------------------------------

typedef __attribute__((ext_vector_type(16))) __bf16        v16bf;
typedef __attribute__((ext_vector_type(8)))  float         v8f;
typedef __attribute__((ext_vector_type(4)))  unsigned int  u32x4;
typedef __attribute__((ext_vector_type(4)))  float         f32x4;

struct U32x8 { u32x4 lo; u32x4 hi; };   // 32B, bit-castable to v16bf

__device__ __forceinline__ unsigned short f32_to_bf16_bits(float f) {
    unsigned int u = __builtin_bit_cast(unsigned int, f);
    u += 0x7FFFu + ((u >> 16) & 1u);          // round-to-nearest-even
    return (unsigned short)(u >> 16);
}

#define BATCH 256
#define SEQ   512
#define EMB   1024
#define UNITS 1024
#define BS    (BATCH * SEQ)   // 131072

// ---------------------------------------------------------------------------
// 0: pack W1 -> bf16 in WMMA B-fragment order, zero logits accumulator.
// Packed layout: tile t = (ct<<5)|kt covers cols [ct*16,+16) x k [kt*32,+32).
// Within a tile, lane l owns 16 bf16: n = ct*16 + (l&15), k = kt*32 + (l>>4)*16 + j.
// Flat: W1p[t*512 + l*16 + j]  ->  each lane loads its 32B with two b128s.
// ---------------------------------------------------------------------------
__global__ void prep_kernel(const float* __restrict__ W1,
                            unsigned short* __restrict__ W1p,
                            float* __restrict__ logits) {
    int g = blockIdx.x * 256 + threadIdx.x;
    if (g < EMB * UNITS) {
        int tile = g >> 9;
        int lane = (g >> 4) & 31;
        int j    = g & 15;
        int ct   = tile >> 5;
        int kt   = tile & 31;
        int n    = ct * 16 + (lane & 15);
        int k    = kt * 32 + (lane >> 4) * 16 + j;
        W1p[g] = f32_to_bf16_bits(W1[(size_t)k * UNITS + n]);
    }
    if (g < BS) logits[g] = 0.0f;
}

// --- 1: ph[b,u] = hidden_prev[b,:] @ W2[:,u] + b2[u] (0.5 GFLOP, trivial) --
__global__ void proj_hidden_kernel(const float* __restrict__ hidden,
                                   const float* __restrict__ W2,
                                   const float* __restrict__ b2,
                                   float* __restrict__ ph) {
    int g = blockIdx.x * 256 + threadIdx.x;      // 256*1024 outputs
    int b = g >> 10, u = g & 1023;
    float acc = b2[u];
    const float* hp = hidden + (size_t)b * 1024;
    for (int h = 0; h < 1024; ++h)
        acc = fmaf(hp[h], W2[(size_t)h * 1024 + u], acc);
    ph[g] = acc;
}

// ---------------------------------------------------------------------------
// 2: fused  pf = feat@W1 ; logits += tanh(pf + b1 + ph) @ Wv
// Block = 8 waves, 64 rows x 128 cols.  Wave (wrow 0..3, wcol 0..1):
// 16 rows x 64 cols = 4 accumulator tiles fed by one shared A fragment.
// A staged in LDS at K=64 depth (one barrier pair per 64 K).
// ---------------------------------------------------------------------------
__global__ __launch_bounds__(256) void score_logits_kernel(
        const float* __restrict__ features,
        const unsigned short* __restrict__ W1p,
        const float* __restrict__ b1,
        const float* __restrict__ ph,
        const float* __restrict__ Wv,
        float* __restrict__ logits) {
    __shared__ __align__(16) unsigned short Alds[64 * 72];   // 144B stride, 16B aligned

    const int tid  = threadIdx.x;
    const int lane = tid & 31;
    const int wave = tid >> 5;
    const int wrow = wave >> 1;            // 0..3
    const int wcol = wave & 1;             // 0..1
    const int lh   = lane >> 4;
    const int l15  = lane & 15;
    const int rowBase = blockIdx.x * 64;               // bs rows
    const int colBase = blockIdx.y * 128 + wcol * 64;  // u cols for this wave
    const int ctgBase = colBase >> 4;                  // global 16-col tile idx

    v8f acc[4] = {};

    for (int k0 = 0; k0 < EMB; k0 += 64) {
        // Stage A: 64 rows x 64 k, f32 -> bf16 (8 float2 per thread, coalesced)
        #pragma unroll
        for (int i = 0; i < 8; ++i) {
            int lin = i * 256 + tid;            // 0..2047 pairs
            int r = lin >> 5, kp = lin & 31;
            const float* src = features + (size_t)(rowBase + r) * EMB + k0 + kp * 2;
            unsigned int pk = (unsigned int)f32_to_bf16_bits(src[0])
                            | ((unsigned int)f32_to_bf16_bits(src[1]) << 16);
            *(unsigned int*)(Alds + r * 72 + kp * 2) = pk;
        }
        if (k0 + 64 < EMB)   // hint next A tile toward the caches
            __builtin_prefetch(features + (size_t)(rowBase + (tid >> 2)) * EMB + k0 + 64, 0, 1);
        __syncthreads();

        #pragma unroll
        for (int kt2 = 0; kt2 < 2; ++kt2) {
            // A frag: lanes 0-15 K{0-15}, lanes 16-31 K{16-31}
            const unsigned short* ap = Alds + (wrow * 16 + l15) * 72 + kt2 * 32;
            U32x8 pk;
            pk.lo = *(const u32x4*)(ap + lh * 16);
            pk.hi = *(const u32x4*)(ap + lh * 16 + 8);
            v16bf afrag = __builtin_bit_cast(v16bf, pk);

            const int ktg = (k0 >> 5) + kt2;
            #pragma unroll
            for (int ct = 0; ct < 4; ++ct) {
                // B frag: pre-packed, 32 contiguous bytes per lane (L2-resident)
                const unsigned short* bp =
                    W1p + (((size_t)(ctgBase + ct) * 32 + ktg) << 9) + lane * 16;
                U32x8 bk;
                bk.lo = *(const u32x4*)(bp);
                bk.hi = *(const u32x4*)(bp + 8);
                v16bf bfrag = __builtin_bit_cast(v16bf, bk);
                acc[ct] = __builtin_amdgcn_wmma_f32_16x16x32_bf16(
                              false, afrag, false, bfrag, (short)0, acc[ct], false, false);
            }
        }
        __syncthreads();
    }

    // Epilogue: p[row] = sum_u tanh(pf + b1 + ph) * Wv over 64 owned cols.
    const int bIdx = rowBase >> 9;                 // batch is block-uniform
    const float* phrow = ph + (size_t)bIdx * UNITS;
    float blv[4], wvv[4], phv[4];
    #pragma unroll
    for (int ct = 0; ct < 4; ++ct) {
        int u = colBase + ct * 16 + l15;
        blv[ct] = b1[u]; wvv[ct] = Wv[u]; phv[ct] = phrow[u];
    }
    #pragma unroll
    for (int r = 0; r < 8; ++r) {
        float p = 0.f;
        #pragma unroll
        for (int ct = 0; ct < 4; ++ct) {
            float pf = acc[ct][r] + blv[ct] + phv[ct];
            p += tanhf(pf) * wvv[ct];
        }
        #pragma unroll
        for (int m = 1; m < 16; m <<= 1) p += __shfl_xor(p, m, 32);
        if (l15 == 0) {
            int row = rowBase + wrow * 16 + r + lh * 8;   // C layout: M = r + 8*half
            atomicAdd(&logits[row], p);
        }
    }
}

// --- 3: softmax over S per batch; attn straight into d_out -----------------
__global__ void softmax_kernel(const float* __restrict__ logits,
                               const float* __restrict__ bv,
                               float* __restrict__ attn) {
    __shared__ float sm[SEQ];
    int b = blockIdx.x, s = threadIdx.x;
    float x = logits[b * SEQ + s] + bv[0];
    sm[s] = x; __syncthreads();
    for (int st = 256; st > 0; st >>= 1) {
        if (s < st) sm[s] = fmaxf(sm[s], sm[s + st]);
        __syncthreads();
    }
    float mx = sm[0]; __syncthreads();
    float e = __expf(x - mx);
    sm[s] = e; __syncthreads();
    for (int st = 256; st > 0; st >>= 1) {
        if (s < st) sm[s] += sm[s + st];
        __syncthreads();
    }
    attn[b * SEQ + s] = e / sm[0];
}

// --- 4: wf[b,e] = sum_s attn[b,s] * features[b,s,e]  (bf16 out) ------------
__global__ void wfeat_kernel(const float* __restrict__ features,
                             const float* __restrict__ attn,
                             unsigned short* __restrict__ wfb) {
    __shared__ float sa[SEQ];
    int b = blockIdx.x, tid = threadIdx.x;
    sa[tid]       = attn[b * SEQ + tid];
    sa[tid + 256] = attn[b * SEQ + tid + 256];
    __syncthreads();
    f32x4 acc = {0.f, 0.f, 0.f, 0.f};
    for (int s = 0; s < SEQ; ++s) {
        float a = sa[s];
        f32x4 f = *(const f32x4*)(features + ((size_t)(b * SEQ + s)) * EMB + tid * 4);
        acc += a * f;
    }
    unsigned short* dst = wfb + (size_t)b * EMB + tid * 4;
    dst[0] = f32_to_bf16_bits(acc.x); dst[1] = f32_to_bf16_bits(acc.y);
    dst[2] = f32_to_bf16_bits(acc.z); dst[3] = f32_to_bf16_bits(acc.w);
}

// --- 5: context = wf @ W1 + b1 (256x1024x1024 WMMA GEMM, packed B) ---------
__global__ __launch_bounds__(256) void context_kernel(
        const unsigned short* __restrict__ wfb,
        const unsigned short* __restrict__ W1p,
        const float* __restrict__ b1,
        float* __restrict__ out) {
    __shared__ __align__(16) unsigned short Alds[64 * 72];

    const int tid = threadIdx.x, lane = tid & 31, wave = tid >> 5;
    const int wrow = wave >> 1, wcol = wave & 1;
    const int lh = lane >> 4, l15 = lane & 15;
    const int rowBase = blockIdx.x * 64;               // batch rows
    const int colBase = blockIdx.y * 128 + wcol * 64;
    const int ctgBase = colBase >> 4;

    v8f acc[4] = {};

    for (int k0 = 0; k0 < EMB; k0 += 64) {
        #pragma unroll
        for (int i = 0; i < 8; ++i) {          // A already bf16: uint copies
            int lin = i * 256 + tid;            // 0..2047 uint pairs
            int r = lin >> 5, kp = lin & 31;
            unsigned int v = *(const unsigned int*)
                (wfb + (size_t)(rowBase + r) * EMB + k0 + kp * 2);
            *(unsigned int*)(Alds + r * 72 + kp * 2) = v;
        }
        __syncthreads();

        #pragma unroll
        for (int kt2 = 0; kt2 < 2; ++kt2) {
            const unsigned short* ap = Alds + (wrow * 16 + l15) * 72 + kt2 * 32;
            U32x8 pk;
            pk.lo = *(const u32x4*)(ap + lh * 16);
            pk.hi = *(const u32x4*)(ap + lh * 16 + 8);
            v16bf afrag = __builtin_bit_cast(v16bf, pk);

            const int ktg = (k0 >> 5) + kt2;
            #pragma unroll
            for (int ct = 0; ct < 4; ++ct) {
                const unsigned short* bp =
                    W1p + (((size_t)(ctgBase + ct) * 32 + ktg) << 9) + lane * 16;
                U32x8 bk;
                bk.lo = *(const u32x4*)(bp);
                bk.hi = *(const u32x4*)(bp + 8);
                v16bf bfrag = __builtin_bit_cast(v16bf, bk);
                acc[ct] = __builtin_amdgcn_wmma_f32_16x16x32_bf16(
                              false, afrag, false, bfrag, (short)0, acc[ct], false, false);
            }
        }
        __syncthreads();
    }

    #pragma unroll
    for (int ct = 0; ct < 4; ++ct) {
        int u = colBase + ct * 16 + l15;
        float bl = b1[u];
        #pragma unroll
        for (int r = 0; r < 8; ++r) {
            int row = rowBase + wrow * 16 + r + lh * 8;
            out[(size_t)row * UNITS + u] = acc[ct][r] + bl;
        }
    }
}

// ---------------------------------------------------------------------------
extern "C" void kernel_launch(void* const* d_in, const int* in_sizes, int n_in,
                              void* d_out, int out_size, void* d_ws, size_t ws_size,
                              hipStream_t stream) {
    const float* features = (const float*)d_in[0];
    const float* hidden   = (const float*)d_in[1];
    const float* W1       = (const float*)d_in[2];
    const float* b1       = (const float*)d_in[3];
    const float* W2       = (const float*)d_in[4];
    const float* b2       = (const float*)d_in[5];
    const float* Wv       = (const float*)d_in[6];
    const float* bv       = (const float*)d_in[7];

    float* out_ctx  = (float*)d_out;             // (256,1024)
    float* out_attn = out_ctx + BATCH * UNITS;   // (256,512,1)

    char* ws = (char*)d_ws;
    unsigned short* W1p    = (unsigned short*)(ws);                         // 2 MB
    float*          ph     = (float*)(ws + (2u << 20));                     // 1 MB
    float*          logits = (float*)(ws + (3u << 20));                     // 512 KB
    unsigned short* wfb    = (unsigned short*)(ws + (3u << 20) + (512u << 10)); // 512 KB

    prep_kernel<<<4096, 256, 0, stream>>>(W1, W1p, logits);
    proj_hidden_kernel<<<1024, 256, 0, stream>>>(hidden, W2, b2, ph);

    dim3 g1(BS / 64, UNITS / 128);               // 2048 x 8
    score_logits_kernel<<<g1, 256, 0, stream>>>(features, W1p, b1, ph, Wv, logits);

    softmax_kernel<<<BATCH, SEQ, 0, stream>>>(logits, bv, out_attn);
    wfeat_kernel<<<BATCH, 256, 0, stream>>>(features, out_attn, wfb);

    dim3 g2(BATCH / 64, UNITS / 128);            // 4 x 8
    context_kernel<<<g2, 256, 0, stream>>>(wfb, W1p, b1, out_ctx);
}
